// MoELayer_67568425500797
// MI455X (gfx1250) — matrix operand order repaired and checked by reading, not compiled, not run on hardware
//
#include <hip/hip_runtime.h>

typedef __attribute__((ext_vector_type(16))) _Float16 v16h;
typedef __attribute__((ext_vector_type(8)))  float    v8f;
typedef __attribute__((ext_vector_type(4)))  float    v4f;

#define D_DIM   256
#define E_NUM   3
#define KSTEPS  (D_DIM / 32)   // 8 WMMA k-steps per 16-token tile
#define NOISE_EPS 0.2f
#define WPACK_ELEMS (KSTEPS * 32 * 16)   // 4096 halves = 8 KB

__device__ __forceinline__ float softplus_f(float x) {
  // numerically stable softplus with fast transcendentals
  return fmaxf(x, 0.0f) + __logf(1.0f + __expf(-fabsf(x)));
}

// One-shot prep: pack A = W^T (rows 0..2 = w_gate cols, 3..5 = w_noise cols,
// 6..15 = zero) into this code's per-lane WMMA A-fragment order (same k order as B):
//   wpack[(kk*32 + lane)*16 + i], lane holds K = kb..kb+15 contiguous (kb = kk*32 + 16*khalf)
__global__ void pack_weights(const float* __restrict__ w_gate,
                             const float* __restrict__ w_noise,
                             _Float16* __restrict__ wpack) {
  const int idx = blockIdx.x * blockDim.x + threadIdx.x;
  if (idx >= WPACK_ELEMS) return;
  const int i     = idx & 15;
  const int lane  = (idx >> 4) & 31;
  const int kk    = idx >> 9;
  const int mrow  = lane & 15;
  const int khalf = lane >> 4;
  const int kb    = kk * 32 + khalf * 16;
  const int k     = kb + i;
  float v = 0.0f;
  if (mrow < E_NUM)          v = w_gate[k * E_NUM + mrow];
  else if (mrow < 2 * E_NUM) v = w_noise[k * E_NUM + (mrow - E_NUM)];
  wpack[idx] = (_Float16)v;
}

// One wave handles a 16-token tile per iteration.
// A = W^T (16 x 256, f16) from the pre-packed L2-resident table.
// B = input^T (256 x 16, f16): token n on lane n, contiguous K per lane.
// C (16x16 f32): lane n (lanes 0..15) holds M=0..7 in c[0..7] -> all 6 logit
// components of token n in-register. No LDS / shuffles needed.
__global__ __launch_bounds__(256) void moe_topk_gate(
    const float* __restrict__ input,
    const _Float16* __restrict__ wpack,
    const float* __restrict__ noise,
    int* __restrict__ out,
    int num_tiles)
{
  const int lane  = threadIdx.x & 31;
  const int wave  = blockIdx.x * (blockDim.x >> 5) + (threadIdx.x >> 5);
  const int total_waves = gridDim.x * (blockDim.x >> 5);
  const int mrow  = lane & 15;   // B,C: token column within tile
  const int khalf = lane >> 4;   // K half selector per ISA fragment layouts

  // ---- Preload A fragments: 8 aligned 32B vector loads (8 KB table, L2-hot) ----
  const v16h* __restrict__ wp = (const v16h*)wpack;
  v16h afrag[KSTEPS];
#pragma unroll
  for (int kk = 0; kk < KSTEPS; ++kk)
    afrag[kk] = wp[kk * 32 + lane];

  // ---- Stream tokens: each input byte read exactly once -> non-temporal ----
  for (int tile = wave; tile < num_tiles; tile += total_waves) {
    // B 32x16, k order matching A: lane holds K = 16*khalf .. +15 contiguous for its token
    const float* rowp = input + (size_t)(tile * 16 + mrow) * D_DIM + khalf * 16;
    v8f c = {};
#pragma unroll
    for (int kk = 0; kk < KSTEPS; ++kk) {
      const v4f* p = (const v4f*)(rowp + kk * 32);
      v4f f0 = __builtin_nontemporal_load(p + 0);
      v4f f1 = __builtin_nontemporal_load(p + 1);
      v4f f2 = __builtin_nontemporal_load(p + 2);
      v4f f3 = __builtin_nontemporal_load(p + 3);
      v16h b;
#pragma unroll
      for (int j = 0; j < 4; ++j) {
        b[j]      = (_Float16)f0[j];
        b[4 + j]  = (_Float16)f1[j];
        b[8 + j]  = (_Float16)f2[j];
        b[12 + j] = (_Float16)f3[j];
      }
      c = __builtin_amdgcn_wmma_f32_16x16x32_f16(
          /*neg_a=*/false, afrag[kk], /*neg_b=*/false, b,
          /*c_mod=*/(short)0, c, /*reuse_a=*/false, /*reuse_b=*/false);
    }

    // ---- Epilogue: lanes 0..15 own tokens; M=0..2 clean, M=3..5 raw noise ----
    if (khalf == 0) {
      const int t = tile * 16 + mrow;
      const float n0 = noise[t * 3 + 0];
      const float n1 = noise[t * 3 + 1];
      const float n2 = noise[t * 3 + 2];
      const float l0 = c[0] + n0 * (softplus_f(c[3]) + NOISE_EPS);
      const float l1 = c[1] + n1 * (softplus_f(c[4]) + NOISE_EPS);
      const float l2 = c[2] + n2 * (softplus_f(c[5]) + NOISE_EPS);
      int idx = 0;
      float best = l0;
      if (l1 > best) { best = l1; idx = 1; }   // strict '>' == jax tie-break (lowest idx)
      if (l2 > best) { idx = 2; }
      out[t] = idx;
    }
  }
}

extern "C" void kernel_launch(void* const* d_in, const int* in_sizes, int n_in,
                              void* d_out, int out_size, void* d_ws, size_t ws_size,
                              hipStream_t stream) {
  const float* input   = (const float*)d_in[0];
  const float* w_gate  = (const float*)d_in[1];
  const float* w_noise = (const float*)d_in[2];
  const float* noise   = (const float*)d_in[3];
  int* out = (int*)d_out;
  _Float16* wpack = (_Float16*)d_ws;       // 8 KB fragment table

  const int N = in_sizes[0] / D_DIM;       // 262144
  const int num_tiles = N / 16;            // 16384 tiles of 16 tokens

  // One-shot weight packing into WMMA fragment layout (graph-capture safe).
  pack_weights<<<(WPACK_ELEMS + 255) / 256, 256, 0, stream>>>(w_gate, w_noise, wpack);

  // 8 waves/block, ~2 tiles per wave: thousands of waves in flight with deep
  // outstanding-load queues to saturate 23.3 TB/s; cheap preload amortizes fine.
  int blocks = (num_tiles + 15) / 16;
  if (blocks > 1024) blocks = 1024;
  if (blocks < 1)    blocks = 1;

  moe_topk_gate<<<blocks, 256, 0, stream>>>(input, wpack, noise, out, num_tiles);
}
